// MambaBlock_73151882985983
// MI455X (gfx1250) — compile-verified, matrix operand order checked by reading, not run on hardware
//
#include <hip/hip_runtime.h>

#define SEQL   2048
#define DMODEL 512
#define DINNER 1024
#define DSTATE 16
#define DTRANK 32
#define NLAYERS 2
#define BATCH  4
#define MROWS  (BATCH * SEQL)   // 8192 token rows

typedef __attribute__((ext_vector_type(16))) _Float16 v16h;
typedef __attribute__((ext_vector_type(8)))  float    v8f;

// ---------------------------------------------------------------------------
// Async copy of 16 bytes global -> LDS (CDNA5 GLOBAL_LOAD_ASYNC_TO_LDS_B128,
// GVS mode: 64-bit SGPR base + 32-bit per-lane byte offset). Tracked by
// ASYNCcnt; no VGPR data movement.
// ---------------------------------------------------------------------------
__device__ __forceinline__ void async_b128(const _Float16* lds_ptr,
                                           const _Float16* gbase,
                                           unsigned goff_bytes) {
  unsigned lds_off = (unsigned)(uintptr_t)lds_ptr;  // low 32 bits = LDS offset
  unsigned long long base64 = (unsigned long long)(uintptr_t)gbase;
  asm volatile("global_load_async_to_lds_b128 %0, %1, %2"
               :
               : "v"(lds_off), "v"(goff_bytes), "s"(base64)
               : "memory");
}
__device__ __forceinline__ void wait_async_le3() {
  asm volatile("s_wait_asynccnt 3" ::: "memory");
}
__device__ __forceinline__ void wait_async_0() {
  asm volatile("s_wait_asynccnt 0" ::: "memory");
}

// ---------------------------------------------------------------------------
// Load a 16x32 f16 WMMA fragment from an LDS tile (row stride 32 halves).
// CDNA5 16-bit A-matrix layout (CDNA5 ISA):
//   lanes 0-15 (half=0): row r, K = 0..7 and 16..23
//   lanes 16-31 (half=1): row r, K = 8..15 and 24..31
// B fragments use the same layout with the tile stored as [N][K].
// ---------------------------------------------------------------------------
__device__ __forceinline__ v16h ld_frag(const _Float16* base, int r, int half) {
  union { v16h v; uint4 u[2]; } f;
  const _Float16* p = base + r * 32 + half * 8;
  f.u[0] = *(const uint4*)(p);        // ds_load_b128
  f.u[1] = *(const uint4*)(p + 16);
  return f.v;
}

// ---------------------------------------------------------------------------
// WMMA GEMM:  C[M,N] = A[M,K] * B[N,K]^T   (A,B f16; f32 accumulate)
// 256 threads = 8 waves; block tile 128(M) x 64(N); K-step 32.
// Double-buffered LDS filled by async-to-LDS; s_wait_asynccnt pipeline.
// ---------------------------------------------------------------------------
template <typename CT>
__global__ __launch_bounds__(256) void gemm_f16_wmma(
    const _Float16* __restrict__ A, int lda,
    const _Float16* __restrict__ B, int ldb,
    CT* __restrict__ C, int ldc, int K) {
  __shared__ __align__(16) _Float16 As[2][128 * 32];
  __shared__ __align__(16) _Float16 Bs[2][64 * 32];

  const int tid  = threadIdx.x;
  const int lane = tid & 31;
  const int wave = tid >> 5;
  const int half = lane >> 4;
  const int r    = lane & 15;
  const int wm   = (wave >> 1) * 32;   // wave M offset in block tile
  const int wn   = (wave & 1) * 32;    // wave N offset in block tile
  const int bM   = blockIdx.y * 128;
  const int bN   = blockIdx.x * 64;
  const int lrow = tid >> 2;           // 0..63
  const int lcol = (tid & 3) * 8;      // 0,8,16,24

  const unsigned aRow0 = ((unsigned)(bM + lrow) * (unsigned)lda + lcol) * 2u;
  const unsigned aRow1 = aRow0 + (unsigned)(64 * lda) * 2u;
  const unsigned bRow  = ((unsigned)(bN + lrow) * (unsigned)ldb + lcol) * 2u;

  v8f acc[2][2] = {};

  // prefetch tile 0
  async_b128(&As[0][lrow * 32 + lcol],        A, aRow0);
  async_b128(&As[0][(lrow + 64) * 32 + lcol], A, aRow1);
  async_b128(&Bs[0][lrow * 32 + lcol],        B, bRow);

  const int nk = K >> 5;
  for (int t = 0; t < nk; ++t) {
    const int cur = t & 1;
    if (t + 1 < nk) {
      const unsigned kb = (unsigned)((t + 1) << 5) * 2u;   // K bytes advance
      async_b128(&As[cur ^ 1][lrow * 32 + lcol],        A, aRow0 + kb);
      async_b128(&As[cur ^ 1][(lrow + 64) * 32 + lcol], A, aRow1 + kb);
      async_b128(&Bs[cur ^ 1][lrow * 32 + lcol],        B, bRow + kb);
      wait_async_le3();   // tile t resident (only the 3 new remain pending)
    } else {
      wait_async_0();
    }
    __syncthreads();

    v16h af[2], bf[2];
    af[0] = ld_frag(&As[cur][(wm +  0) * 32], r, half);
    af[1] = ld_frag(&As[cur][(wm + 16) * 32], r, half);
    bf[0] = ld_frag(&Bs[cur][(wn +  0) * 32], r, half);
    bf[1] = ld_frag(&Bs[cur][(wn + 16) * 32], r, half);

#pragma unroll
    for (int i = 0; i < 2; ++i)
#pragma unroll
      for (int j = 0; j < 2; ++j)
        acc[i][j] = __builtin_amdgcn_wmma_f32_16x16x32_f16(
            false, af[i], false, bf[j], (short)0, acc[i][j], false, false);
    __syncthreads();   // tile `cur` free for reuse at iteration t+2
  }

  // C/D layout: VGPR v holds (M = v + half*8, N = r) within each 16x16 tile.
#pragma unroll
  for (int i = 0; i < 2; ++i)
#pragma unroll
    for (int j = 0; j < 2; ++j)
#pragma unroll
      for (int v = 0; v < 8; ++v) {
        int row = bM + wm + i * 16 + half * 8 + v;
        int col = bN + wn + j * 16 + r;
        C[(size_t)row * ldc + col] = (CT)acc[i][j][v];
      }
}

// ---------------------------------------------------------------------------
// LayerNorm over d_model=512 -> f16 output. One block (256 thr) per row.
// ---------------------------------------------------------------------------
__global__ __launch_bounds__(256) void layernorm_f16_kernel(
    const float* __restrict__ x, const float* __restrict__ w,
    const float* __restrict__ b, _Float16* __restrict__ out) {
  __shared__ float sh[256], sh2[256];
  const int row = blockIdx.x;
  const int tid = threadIdx.x;
  const float* xr = x + (size_t)row * DMODEL;
  float v0 = xr[tid], v1 = xr[tid + 256];
  sh[tid]  = v0 + v1;
  sh2[tid] = v0 * v0 + v1 * v1;
  __syncthreads();
  for (int st = 128; st > 0; st >>= 1) {
    if (tid < st) { sh[tid] += sh[tid + st]; sh2[tid] += sh2[tid + st]; }
    __syncthreads();
  }
  float mean = sh[0] * (1.0f / DMODEL);
  float var  = sh2[0] * (1.0f / DMODEL) - mean * mean;
  float rstd = rsqrtf(var + 1e-5f);
  out[(size_t)row * DMODEL + tid]       = (_Float16)((v0 - mean) * rstd * w[tid] + b[tid]);
  out[(size_t)row * DMODEL + tid + 256] = (_Float16)((v1 - mean) * rstd * w[tid + 256] + b[tid + 256]);
}

__global__ void cvt_f32_f16(const float* __restrict__ s, _Float16* __restrict__ d, int n) {
  int i = blockIdx.x * 256 + threadIdx.x;
  if (i < n) d[i] = (_Float16)s[i];
}

// Causal depthwise conv1d (width 4) + bias + SiLU. x = first half of xz.
__global__ __launch_bounds__(256) void conv_silu_kernel(
    const _Float16* __restrict__ xz, const float* __restrict__ cw,
    const float* __restrict__ cb, _Float16* __restrict__ xc) {
  int idx = blockIdx.x * 256 + threadIdx.x;   // over MROWS*DINNER
  int c = idx & (DINNER - 1);
  int m = idx >> 10;
  int l = m & (SEQL - 1);
  float acc = cb[c];
  const float* w = cw + c * 4;
#pragma unroll
  for (int j = 0; j < 4; ++j) {
    int ll = l - 3 + j;
    if (ll >= 0)
      acc += w[j] * (float)xz[(size_t)(m - 3 + j) * (2 * DINNER) + c];
  }
  float s = acc / (1.0f + __expf(-acc));
  xc[idx] = (_Float16)s;
}

// delta = softplus(dt_proj_out + dt_bias), in place on f16 buffer.
__global__ void bias_softplus_kernel(_Float16* __restrict__ d,
                                     const float* __restrict__ bias) {
  int idx = blockIdx.x * 256 + threadIdx.x;
  int c = idx & (DINNER - 1);
  float v = (float)d[idx] + bias[c];
  float sp = (v > 20.f) ? v : log1pf(__expf(v));
  d[idx] = (_Float16)sp;
}

// ---------------------------------------------------------------------------
// Selective scan: one thread per (batch, channel), 16-state recurrence in
// registers, serial over L. B/C rows for 16 timesteps staged in LDS per
// chunk (broadcast reads). Fuses  + x*D  and  * silu(z).
// ---------------------------------------------------------------------------
__global__ __launch_bounds__(256) void scan_kernel(
    const _Float16* __restrict__ delta, const _Float16* __restrict__ x,
    const _Float16* __restrict__ xdbl, const _Float16* __restrict__ xz,
    const float* __restrict__ A_log, const float* __restrict__ Dp,
    _Float16* __restrict__ y) {
  __shared__ __align__(8) _Float16 bcs[16][64];   // 16 timesteps of [dt|B|C]
  const int d = blockIdx.x * 256 + threadIdx.x;   // 0..1023
  const int b = blockIdx.y;
  const int tid = threadIdx.x;
  float a[DSTATE], s[DSTATE];
#pragma unroll
  for (int n = 0; n < DSTATE; ++n) {
    a[n] = -__expf(A_log[d * DSTATE + n]);
    s[n] = 0.0f;
  }
  const float Dv = Dp[d];
  const size_t base = (size_t)b * SEQL;
  const int srow = tid >> 4;          // 0..15
  const int scol = (tid & 15) * 4;    // 0..60 halves (8B chunks)

  for (int l0 = 0; l0 < SEQL; l0 += 16) {
    // stage 16 rows x 64 halves of xdbl into LDS (one uint2 per thread)
    *(uint2*)&bcs[srow][scol] =
        *(const uint2*)&xdbl[(base + l0 + srow) * 64 + scol];
    if (l0 + 16 < SEQL) {   // prefetch next chunk of the streaming operands
      __builtin_prefetch(&delta[(base + l0 + 16) * DINNER + d], 0, 1);
      __builtin_prefetch(&x[(base + l0 + 16) * DINNER + d], 0, 1);
    }
    __syncthreads();
#pragma unroll 4
    for (int li = 0; li < 16; ++li) {
      const size_t m = base + l0 + li;
      float dt = (float)delta[m * DINNER + d];
      float xv = (float)x[m * DINNER + d];
      float bu = dt * xv;
      float yv = 0.0f;
#pragma unroll
      for (int n = 0; n < DSTATE; ++n) {
        float dA = __expf(dt * a[n]);
        s[n] = dA * s[n] + bu * (float)bcs[li][DTRANK + n];
        yv += s[n] * (float)bcs[li][DTRANK + DSTATE + n];
      }
      yv += xv * Dv;
      float zv = (float)xz[m * (2 * DINNER) + DINNER + d];
      yv *= zv / (1.0f + __expf(-zv));
      y[m * DINNER + d] = (_Float16)yv;
    }
    __syncthreads();
  }
}

__global__ void add_kernel(const float* __restrict__ a, const float* __restrict__ b,
                           float* __restrict__ o, int n) {
  int i = blockIdx.x * 256 + threadIdx.x;
  if (i < n) o[i] = a[i] + b[i];
}

// ---------------------------------------------------------------------------
extern "C" void kernel_launch(void* const* d_in, const int* in_sizes, int n_in,
                              void* d_out, int out_size, void* d_ws, size_t ws_size,
                              hipStream_t stream) {
  const float* h       = (const float*)d_in[0];
  const float* norm_w  = (const float*)d_in[1];
  const float* norm_b  = (const float*)d_in[2];
  const float* in_w    = (const float*)d_in[3];
  const float* conv_w  = (const float*)d_in[4];
  const float* conv_b  = (const float*)d_in[5];
  const float* xproj_w = (const float*)d_in[6];
  const float* dt_w    = (const float*)d_in[7];
  const float* dt_b    = (const float*)d_in[8];
  const float* A_log   = (const float*)d_in[9];
  const float* Dp      = (const float*)d_in[10];
  const float* out_w   = (const float*)d_in[11];
  (void)in_sizes; (void)n_in; (void)ws_size;

  char* ws = (char*)d_ws;
  size_t off = 0;
  auto alloc = [&](size_t bytes) {
    char* p = ws + off;
    off += (bytes + 255) & ~(size_t)255;
    return p;
  };
  _Float16* hn16   = (_Float16*)alloc((size_t)MROWS * DMODEL * 2);
  _Float16* xz16   = (_Float16*)alloc((size_t)MROWS * 2 * DINNER * 2);
  _Float16* xc16   = (_Float16*)alloc((size_t)MROWS * DINNER * 2);
  _Float16* xdbl16 = (_Float16*)alloc((size_t)MROWS * 64 * 2);
  _Float16* del16  = (_Float16*)alloc((size_t)MROWS * DINNER * 2);
  _Float16* y16    = (_Float16*)alloc((size_t)MROWS * DINNER * 2);
  float*    hcur   = (float*)   alloc((size_t)MROWS * DMODEL * 4);
  _Float16* w_in16 = (_Float16*)alloc((size_t)2 * DINNER * DMODEL * 2);
  _Float16* w_x16  = (_Float16*)alloc((size_t)64 * DINNER * 2);
  _Float16* w_dt16 = (_Float16*)alloc((size_t)DINNER * DTRANK * 2);
  _Float16* w_o16  = (_Float16*)alloc((size_t)DMODEL * DINNER * 2);

  const float* cur = h;
  for (int l = 0; l < NLAYERS; ++l) {
    const int nIn = 2 * DINNER * DMODEL;
    const int nX  = 64 * DINNER;
    const int nDt = DINNER * DTRANK;
    const int nO  = DMODEL * DINNER;
    cvt_f32_f16<<<(nIn + 255) / 256, 256, 0, stream>>>(in_w   + (size_t)l * nIn, w_in16, nIn);
    cvt_f32_f16<<<(nX  + 255) / 256, 256, 0, stream>>>(xproj_w+ (size_t)l * nX,  w_x16,  nX);
    cvt_f32_f16<<<(nDt + 255) / 256, 256, 0, stream>>>(dt_w   + (size_t)l * nDt, w_dt16, nDt);
    cvt_f32_f16<<<(nO  + 255) / 256, 256, 0, stream>>>(out_w  + (size_t)l * nO,  w_o16,  nO);

    layernorm_f16_kernel<<<MROWS, 256, 0, stream>>>(
        cur, norm_w + l * DMODEL, norm_b + l * DMODEL, hn16);

    // in_proj: [8192,512] x [2048,512]^T -> xz [8192,2048]
    gemm_f16_wmma<_Float16><<<dim3((2 * DINNER) / 64, MROWS / 128), 256, 0, stream>>>(
        hn16, DMODEL, w_in16, DMODEL, xz16, 2 * DINNER, DMODEL);

    conv_silu_kernel<<<((size_t)MROWS * DINNER) / 256, 256, 0, stream>>>(
        xz16, conv_w + (size_t)l * DINNER * 4, conv_b + (size_t)l * DINNER, xc16);

    // x_proj: [8192,1024] x [64,1024]^T -> xdbl [8192,64]
    gemm_f16_wmma<_Float16><<<dim3(64 / 64, MROWS / 128), 256, 0, stream>>>(
        xc16, DINNER, w_x16, DINNER, xdbl16, 64, DINNER);

    // dt_proj: [8192,32] x [1024,32]^T -> del [8192,1024]  (A = dt cols of xdbl)
    gemm_f16_wmma<_Float16><<<dim3(DINNER / 64, MROWS / 128), 256, 0, stream>>>(
        xdbl16, 64, w_dt16, DTRANK, del16, DINNER, DTRANK);

    bias_softplus_kernel<<<((size_t)MROWS * DINNER) / 256, 256, 0, stream>>>(
        del16, dt_b + (size_t)l * DINNER);

    scan_kernel<<<dim3(DINNER / 256, BATCH), 256, 0, stream>>>(
        del16, xc16, xdbl16, xz16,
        A_log + (size_t)l * DINNER * DSTATE, Dp + (size_t)l * DINNER, y16);

    // out_proj: [8192,1024] x [512,1024]^T -> hcur [8192,512] (f32)
    gemm_f16_wmma<float><<<dim3(DMODEL / 64, MROWS / 128), 256, 0, stream>>>(
        y16, DINNER, w_o16, DINNER, hcur, DMODEL, DINNER);
    cur = hcur;
  }

  const int nOut = MROWS * DMODEL;
  add_kernel<<<(nOut + 255) / 256, 256, 0, stream>>>(hcur, h, (float*)d_out, nOut);
}
